// GeoFlowNet_84920093377373
// MI455X (gfx1250) — matrix operand order repaired and checked by reading, not compiled, not run on hardware
//
#include <hip/hip_runtime.h>
#include <math.h>

typedef __attribute__((ext_vector_type(16))) _Float16 v16h;
typedef __attribute__((ext_vector_type(8)))  float    v8f;
typedef __attribute__((ext_vector_type(4)))  float    f32x4;
typedef __attribute__((ext_vector_type(2)))  float    v2f;
typedef __attribute__((ext_vector_type(4)))  unsigned int u32x4;
typedef __attribute__((ext_vector_type(8)))  unsigned int u32x8;

// ---------------------------------------------------------------------------
// Tensor Data Mover: 2-D tile (nrows x rowlen f32, row stride stride_elts)
// from global memory into LDS at lds_off. D# groups per CDNA5 ISA:
// group0: count=1 | lds_addr | global_addr(57b) | type=2
// group1: data_size=4B | tensor_dim0/1 | tile_dim0/1 | dim0_stride
// 2-D tensors need only VADDR0(4 SGPRs) + VADDR1(8 SGPRs).
// ---------------------------------------------------------------------------
__device__ __forceinline__ void tdm_load_2d(const float* gptr, unsigned lds_off,
                                            unsigned rowlen, unsigned nrows,
                                            unsigned stride_elts) {
  unsigned long long ga = (unsigned long long)(uintptr_t)gptr;
  u32x4 g0;
  g0[0] = 1u;                                            // count=1, user D#
  g0[1] = lds_off;                                       // lds_addr
  g0[2] = (unsigned)ga;                                  // global_addr lo
  g0[3] = (unsigned)((ga >> 32) & 0x1FFFFFFu) | (2u << 30); // addr hi | type=2
  u32x8 g1;
  g1[0] = (2u << 16);                                    // data_size = 4B
  g1[1] = rowlen << 16;                                  // tensor_dim0 lo16
  g1[2] = nrows << 16;                                   // dim0 hi | tensor_dim1 lo
  g1[3] = rowlen << 16;                                  // dim1 hi | tile_dim0
  g1[4] = nrows;                                         // tile_dim1 | tile_dim2=0
  g1[5] = stride_elts;                                   // tensor_dim0_stride lo
  g1[6] = 0u;                                            // stride hi | dim1_stride lo
  g1[7] = 0u;
  asm volatile("tensor_load_to_lds %0, %1" :: "s"(g0), "s"(g1) : "memory");
}

// ---------------------------------------------------------------------------
// Furthest-point sampling: one workgroup, distance array LDS-resident.
// Matches jnp.argmax first-index tie-breaking.
// ---------------------------------------------------------------------------
__global__ void fps_kernel(const float* __restrict__ pts, int n, int npts,
                           float* __restrict__ out) {
  __shared__ float dc[4096];
  __shared__ float rv[256];
  __shared__ int   ri[256];
  __shared__ float cur[3];
  const int tid = threadIdx.x;

  float p0x = pts[0], p0y = pts[1], p0z = pts[2];
  for (int j = tid; j < n; j += blockDim.x) {
    float dx = pts[3*j+0]-p0x, dy = pts[3*j+1]-p0y, dz = pts[3*j+2]-p0z;
    dc[j] = dx*dx + dy*dy + dz*dz;
  }
  if (tid == 0) { out[0]=p0x; out[1]=p0y; out[2]=p0z; }
  __syncthreads();

  for (int s = 1; s < npts; ++s) {
    float bv = -1.0f; int bi = 0;
    for (int j = tid; j < n; j += blockDim.x) {
      float v = dc[j];
      if (v > bv) { bv = v; bi = j; }       // ascending scan: first max wins
    }
    rv[tid] = bv; ri[tid] = bi;
    __syncthreads();
    for (int off = blockDim.x >> 1; off > 0; off >>= 1) {
      if (tid < off) {
        float v2 = rv[tid+off]; int i2 = ri[tid+off];
        if (v2 > rv[tid] || (v2 == rv[tid] && i2 < ri[tid])) { rv[tid]=v2; ri[tid]=i2; }
      }
      __syncthreads();
    }
    if (tid == 0) {
      int i = ri[0];
      cur[0]=pts[3*i]; cur[1]=pts[3*i+1]; cur[2]=pts[3*i+2];
      out[3*s]=cur[0]; out[3*s+1]=cur[1]; out[3*s+2]=cur[2];
    }
    __syncthreads();
    float cx=cur[0], cy=cur[1], cz=cur[2];
    for (int j = tid; j < n; j += blockDim.x) {
      float dx = pts[3*j]-cx, dy = pts[3*j+1]-cy, dz = pts[3*j+2]-cz;
      dc[j] = fminf(dc[j], dx*dx + dy*dy + dz*dz);
    }
    __syncthreads();
  }
}

// ---------------------------------------------------------------------------
// Linear (+bias, optional ReLU) with up-to-3-way channel concat on the input.
// One wave per 16x16 output tile; v_wmma_f32_16x16x32_f16, f32 accumulate.
// All fragment loads are BRANCHLESS: addresses for all concat segments are
// computed unconditionally and selected with integer cndmask, so the 16 loads
// per fragment issue as one clause under a single staggered wait.
// Fast path: non-concat, K%8==0 -> contiguous b128 loads for A.
// ---------------------------------------------------------------------------
__global__ void linear_kernel(const float* __restrict__ x0, int c0,
                              const float* __restrict__ x1, int c1,
                              const float* __restrict__ x2, int c2,
                              const float* __restrict__ W,
                              const float* __restrict__ bias,
                              float* __restrict__ out,
                              int M, int K, int Nout, int relu) {
  const int lane = threadIdx.x & 31;
  const int hi   = lane >> 4;
  const int lx   = lane & 15;
  const int m0   = blockIdx.x * 16;
  const int n0   = blockIdx.y * 16;
  const int row  = m0 + lx;
  const int col  = n0 + lx;
  const bool fast = (c1 == 0) && (c2 == 0) && ((K & 7) == 0);

  // per-row segment base addresses (computed once, selected per element)
  const uintptr_t b0 = (uintptr_t)(x0 + (size_t)row * c0);
  const uintptr_t b1 = (uintptr_t)(x1 + (size_t)row * c1) - (uintptr_t)c0 * 4u;
  const uintptr_t b2 = (uintptr_t)(x2 + (size_t)row * c2) - (uintptr_t)(c0 + c1) * 4u;

  v8f acc = {};
  for (int k0 = 0; k0 < K; k0 += 32) {
    v16h a, b;
    if (fast) {
      const float* rp = x0 + (size_t)row * K;
#pragma unroll
      for (int cb = 0; cb < 2; ++cb) {              // two aligned 8-wide chunks
        int kb  = k0 + (cb << 4) + (hi << 3);
        bool ok = (kb + 8) <= K;
        int kbc = ok ? kb : 0;
        f32x4 q0 = *(const f32x4*)(rp + kbc);
        f32x4 q1 = *(const f32x4*)(rp + kbc + 4);
#pragma unroll
        for (int j = 0; j < 4; ++j) {
          a[cb*8 + j]     = ok ? (_Float16)q0[j] : (_Float16)0.0f;
          a[cb*8 + 4 + j] = ok ? (_Float16)q1[j] : (_Float16)0.0f;
        }
      }
    } else {
#pragma unroll
      for (int h = 0; h < 16; ++h) {
        int k = k0 + ((h >> 3) << 4) + (hi << 3) + (h & 7);
        bool inb = k < K;
        int kc = inb ? k : 0;
        // integer address select -> v_cndmask, no exec-mask branches
        uintptr_t base = (kc >= c0 + c1) ? b2 : ((kc >= c0) ? b1 : b0);
        float v = *(const float*)(base + (uintptr_t)kc * 4u);
        a[h] = inb ? (_Float16)v : (_Float16)0.0f;
      }
    }
#pragma unroll
    for (int h = 0; h < 16; ++h) {
      int k = k0 + ((h >> 3) << 4) + (hi << 3) + (h & 7);
      bool inb = (k < K) & (col < Nout);
      int kc = (k   < K)    ? k   : 0;
      int cc = (col < Nout) ? col : 0;
      float v = W[kc * Nout + cc];
      b[h] = inb ? (_Float16)v : (_Float16)0.0f;
    }
    acc = __builtin_amdgcn_wmma_f32_16x16x32_f16(false, a, false, b,
                                                 (short)0, acc, false, false);
  }
  if (col < Nout) {
    float bb = bias[col];
#pragma unroll
    for (int q = 0; q < 8; ++q) {
      int r = m0 + q + (hi << 3);
      float y = acc[q] + bb;
      if (relu) y = fmaxf(y, 0.0f);
      out[r * Nout + col] = y;
    }
  }
}

// ---------------------------------------------------------------------------
// Fused multi-radius Gaussian spatial conv (flash style) with TDM streaming.
// One wave per 16-target tile, all C channels. For each 32-wide source tile:
//   - feature tile (32 x C f32) is DMA'd global->LDS by the Tensor Data Mover,
//     double-buffered (issue tile i+1, s_wait_tensorcnt for tile i)
//   - d2 tile via v_wmma_f32_16x16x4_f32 (coords stay f32) -> LDS, read once
//   - per radius: exp on VALU (co-issues with XDL), f16 A-fragment,
//     row-sum via shfl_xor(16), numerator via v_wmma_f32_16x16x32_f16
// Nothing O(Nt*Ns) ever touches global memory.
// ---------------------------------------------------------------------------
template <int C>
__global__ void sconv_kernel(const float* __restrict__ tgt,
                             const float* __restrict__ src,
                             const float* __restrict__ fea,
                             float* __restrict__ out,
                             int Nt, int Ns,
                             float i2r0, float i2r1, float i2r2) {
  constexpr int CT = C / 16;
  const int lane = threadIdx.x & 31;
  const int hi   = lane >> 4;
  const int lx   = lane & 15;
  const int m0   = blockIdx.x * 16;

  __shared__ float tx[16], ty[16], tz[16], t2s[16];
  __shared__ float sx[32], sy[32], sz[32], s2s[32];
  __shared__ float d2t[16][32];
  __shared__ float feaS[2][32 * C];                 // TDM double buffer

  if (lane < 16) {
    float x = tgt[3*(m0+lane)+0], y = tgt[3*(m0+lane)+1], z = tgt[3*(m0+lane)+2];
    tx[lane]=x; ty[lane]=y; tz[lane]=z; t2s[lane]=x*x+y*y+z*z;
  }
  __syncthreads();

  const float i2r[3] = {i2r0, i2r1, i2r2};
  v8f acc[3][CT];
#pragma unroll
  for (int r = 0; r < 3; ++r)
#pragma unroll
    for (int t = 0; t < CT; ++t) acc[r][t] = v8f{};
  float denom[3] = {0.0f, 0.0f, 0.0f};

#if __has_builtin(__builtin_amdgcn_wmma_f32_16x16x4_f32)
  v2f ta;                                           // A frag: (x,y) | (z,0)
  ta[0] = hi ? tz[lx] : tx[lx];
  ta[1] = hi ? 0.0f   : ty[lx];
#endif

  // kick off DMA of the first feature tile
  tdm_load_2d(fea, (unsigned)(uintptr_t)&feaS[0][0], C, 32, C);

  for (int s0 = 0; s0 < Ns; s0 += 32) {
    const int buf = (s0 >> 5) & 1;
    {
      int si = s0 + lane;
      float x = src[3*si+0], y = src[3*si+1], z = src[3*si+2];
      sx[lane]=x; sy[lane]=y; sz[lane]=z; s2s[lane]=x*x+y*y+z*z;
    }
    if (s0 + 32 < Ns) __builtin_prefetch(&src[3*(s0+32)], 0, 1);
    __syncthreads();

#if __has_builtin(__builtin_amdgcn_wmma_f32_16x16x4_f32)
#pragma unroll
    for (int sub = 0; sub < 2; ++sub) {
      int nl = sub*16 + lx;
      v2f sb;
      sb[0] = hi ? sz[nl] : sx[nl];
      sb[1] = hi ? 0.0f   : sy[nl];
      v8f dot = {};
      dot = __builtin_amdgcn_wmma_f32_16x16x4_f32(false, ta, false, sb,
                                                  (short)0, dot, false, false);
#pragma unroll
      for (int q = 0; q < 8; ++q) {
        int m = q + (hi << 3);
        d2t[m][nl] = fmaxf(t2s[m] + s2s[nl] - 2.0f * dot[q], 0.0f);
      }
    }
#else
#pragma unroll
    for (int sub = 0; sub < 2; ++sub) {
      int nl = sub*16 + lx;
#pragma unroll
      for (int q = 0; q < 8; ++q) {
        int m = q + (hi << 3);
        float dx = tx[m]-sx[nl], dy = ty[m]-sy[nl], dz = tz[m]-sz[nl];
        d2t[m][nl] = dx*dx + dy*dy + dz*dz;
      }
    }
#endif

    // issue next tile's DMA, then wait for the current tile (in-order TENSORcnt)
    if (s0 + 32 < Ns) {
      tdm_load_2d(fea + (size_t)(s0 + 32) * C,
                  (unsigned)(uintptr_t)&feaS[buf ^ 1][0], C, 32, C);
      __builtin_amdgcn_s_wait_tensorcnt(1);
    } else {
      __builtin_amdgcn_s_wait_tensorcnt(0);
    }
    asm volatile("" ::: "memory");
    __syncthreads();

    // feature B fragments from LDS, shared across the 3 radii
    const float* fs = &feaS[buf][0];
    v16h feaB[CT];
#pragma unroll
    for (int t = 0; t < CT; ++t) {
      int col = t*16 + lx;
#pragma unroll
      for (int h = 0; h < 16; ++h) {
        int srow = ((h >> 3) << 4) + (hi << 3) + (h & 7);
        feaB[t][h] = (_Float16)fs[srow * C + col];
      }
    }

    // d2 row read once, reused across radii (two contiguous 8-float runs)
    float d2r[16];
#pragma unroll
    for (int h = 0; h < 16; ++h) {
      int kl = ((h >> 3) << 4) + (hi << 3) + (h & 7);
      d2r[h] = d2t[lx][kl];
    }

#pragma unroll
    for (int r = 0; r < 3; ++r) {
      v16h kf;
      float rs = 0.0f;
#pragma unroll
      for (int h = 0; h < 16; ++h) {
        float e = __expf(-d2r[h] * i2r[r]);
        kf[h] = (_Float16)e;
        rs += e;
      }
      rs += __shfl_xor(rs, 16, 32);   // lane L and L^16 hold halves of row lx
      denom[r] += rs;
#pragma unroll
      for (int t = 0; t < CT; ++t)
        acc[r][t] = __builtin_amdgcn_wmma_f32_16x16x32_f16(
            false, kf, false, feaB[t], (short)0, acc[r][t], false, false);
    }
    __syncthreads();
  }

#pragma unroll
  for (int t = 0; t < CT; ++t) {
    int col = t*16 + lx;
#pragma unroll
    for (int q = 0; q < 8; ++q) {
      int m = q + (hi << 3);
      float o = 0.0f;
#pragma unroll
      for (int r = 0; r < 3; ++r) {
        float dm = __shfl(denom[r], m, 32);   // row sums live in lane m
        o += 0.33f * acc[r][t][q] / (dm + 1e-8f);
      }
      out[(size_t)(m0 + m) * C + col] = o;
    }
  }
}

// ---------------------------------------------------------------------------
// Host orchestration
// ---------------------------------------------------------------------------
extern "C" void kernel_launch(void* const* d_in, const int* in_sizes, int n_in,
                              void* d_out, int out_size, void* d_ws, size_t ws_size,
                              hipStream_t stream) {
  const float* pc1 = (const float*)d_in[0];
  const float* pc2 = (const float*)d_in[1];
  const float* f1  = (const float*)d_in[2];
  const float* f2  = (const float*)d_in[3];

  enum { CC0_0=0, CC0_1, CC0_2, CC1_0, CC1_1, CC1_2, CC2_0, CC2_1, CC2_2,
         CC2_PC2, CC3_0, CC3_1, CC3_2, CC2_3, CC2_4, CC2_5, CC1_3, CC1_4,
         CC1_5, CC0_3, CC0_4, CC0_5, REFINE };
  auto Wp = [&](int i){ return (const float*)d_in[4 + 2*i]; };
  auto Bp = [&](int i){ return (const float*)d_in[5 + 2*i]; };

  char* wsp = (char*)d_ws;
  auto alloc = [&](size_t nf){ float* p = (float*)wsp; wsp += nf*sizeof(float); return p; };

  auto lin = [&](const float* x0,int c0,const float* x1,int c1,
                 const float* x2,int c2,int pi,float* y,int M,int N,int relu){
    int K = c0 + c1 + c2;
    dim3 g(M/16, (N+15)/16);
    linear_kernel<<<g, 32, 0, stream>>>(x0,c0,x1,c1,x2,c2,Wp(pi),Bp(pi),y,M,K,N,relu);
  };
  auto lin1 = [&](const float* x,int K,int pi,float* y,int M,int N,int relu){
    lin(x,K,nullptr,0,nullptr,0,pi,y,M,N,relu);
  };
  auto sconv = [&](const float* t,int Nt,const float* s,int Ns,
                   const float* fe,int C,float* y,const float* i2){
    dim3 g(Nt/16);
    switch (C) {
      case 16: sconv_kernel<16><<<g,32,0,stream>>>(t,s,fe,y,Nt,Ns,i2[0],i2[1],i2[2]); break;
      case 32: sconv_kernel<32><<<g,32,0,stream>>>(t,s,fe,y,Nt,Ns,i2[0],i2[1],i2[2]); break;
      case 64: sconv_kernel<64><<<g,32,0,stream>>>(t,s,fe,y,Nt,Ns,i2[0],i2[1],i2[2]); break;
      case 96: sconv_kernel<96><<<g,32,0,stream>>>(t,s,fe,y,Nt,Ns,i2[0],i2[1],i2[2]); break;
    }
  };

  // i2r = 1/(2 r^2) per radius set: R0=_rl(.05), R1=_rl(.2), R2=_rl(.8), R3=_rl(1.6)
  const float i2R0[3] = {8.0f, 2.0f, 0.5f};
  const float i2R1[3] = {0.5f, 0.125f, 0.03125f};
  const float i2R2[3] = {0.03125f, 0.0078125f, 1.0f/512.0f};
  const float i2R3[3] = {0.0078125f, 1.0f/512.0f, 1.0f/2048.0f};

  const int N0 = 4096, N1 = 1024, N2 = 256, N3 = 128;

  // ---- level 0 encoder (both branches) ----
  float* tA   = alloc(N0*8);
  float* tB   = alloc(N0*16);
  float* sA   = alloc(N0*16);
  float* l0f1 = alloc(N0*16);
  float* l0f2 = alloc(N0*16);

  lin1(f1, 1, CC0_0, tA, N0, 8, 1);
  lin1(tA, 8, CC0_1, tB, N0, 16, 1);
  sconv(pc1, N0, pc1, N0, tB, 16, sA, i2R0);
  lin1(sA, 16, CC0_2, l0f1, N0, 16, 1);

  lin1(f2, 1, CC0_0, tA, N0, 8, 1);
  lin1(tA, 8, CC0_1, tB, N0, 16, 1);
  sconv(pc2, N0, pc2, N0, tB, 16, sA, i2R0);
  lin1(sA, 16, CC0_2, l0f2, N0, 16, 1);

  // ---- level 1 ----
  float* l1p1 = alloc(N1*3);
  float* l1p2 = alloc(N1*3);
  fps_kernel<<<1, 256, 0, stream>>>(pc1, N0, N1, l1p1);
  fps_kernel<<<1, 256, 0, stream>>>(pc2, N0, N1, l1p2);

  float* s1   = alloc(N1*16);
  float* t1a  = alloc(N1*32);
  float* t1b  = alloc(N1*32);
  float* s1b  = alloc(N1*32);
  float* l1f1 = alloc(N1*32);
  float* l1f2 = alloc(N1*32);

  sconv(l1p1, N1, pc1, N0, l0f1, 16, s1, i2R1);
  lin1(s1, 16, CC1_0, t1a, N1, 32, 1);
  lin1(t1a, 32, CC1_1, t1b, N1, 32, 1);
  sconv(l1p1, N1, l1p1, N1, t1b, 32, s1b, i2R1);
  lin1(s1b, 32, CC1_2, l1f1, N1, 32, 1);

  sconv(l1p2, N1, pc2, N0, l0f2, 16, s1, i2R1);
  lin1(s1, 16, CC1_0, t1a, N1, 32, 1);
  lin1(t1a, 32, CC1_1, t1b, N1, 32, 1);
  sconv(l1p2, N1, l1p2, N1, t1b, 32, s1b, i2R1);
  lin1(s1b, 32, CC1_2, l1f2, N1, 32, 1);

  // ---- level 2 ----
  float* l2p1 = alloc(N2*3);
  float* l2p2 = alloc(N2*3);
  fps_kernel<<<1, 256, 0, stream>>>(l1p1, N1, N2, l2p1);
  fps_kernel<<<1, 256, 0, stream>>>(l1p2, N1, N2, l2p2);

  float* s2    = alloc(N2*32);
  float* t2a   = alloc(N2*64);
  float* l2f1_ = alloc(N2*64);
  float* l2f2_ = alloc(N2*64);

  sconv(l2p1, N2, l1p1, N1, l1f1, 32, s2, i2R2);
  lin1(s2, 32, CC2_0, t2a, N2, 64, 1);
  lin1(t2a, 64, CC2_1, l2f1_, N2, 64, 1);
  sconv(l2p2, N2, l1p2, N1, l1f2, 32, s2, i2R2);
  lin1(s2, 32, CC2_0, t2a, N2, 64, 1);
  lin1(t2a, 64, CC2_1, l2f2_, N2, 64, 1);

  float* s2x  = alloc(N2*64);
  float* l2f1 = alloc(N2*64);
  float* l2f2 = alloc(N2*64);
  sconv(l2p1, N2, l2p2, N2, l2f2_, 64, s2x, i2R2);      // cross-cloud correlation
  lin(s2x, 64, l2f1_, 64, nullptr, 0, CC2_2, l2f1, N2, 64, 1);
  lin1(l2f2_, 64, CC2_PC2, l2f2, N2, 64, 1);

  // ---- level 3 ----
  float* l3p1 = alloc(N3*3);
  float* l3p2 = alloc(N3*3);
  fps_kernel<<<1, 256, 0, stream>>>(l2p1, N2, N3, l3p1);
  fps_kernel<<<1, 256, 0, stream>>>(l2p2, N2, N3, l3p2);

  float* s3    = alloc(N3*64);
  float* t3    = alloc(N3*64);
  float* l3f1_ = alloc(N3*64);
  float* l3f2_ = alloc(N3*64);
  sconv(l3p1, N3, l2p1, N2, l2f1, 64, s3, i2R3);
  lin1(s3, 64, CC3_0, t3, N3, 64, 1);
  lin1(t3, 64, CC3_1, l3f1_, N3, 64, 1);
  sconv(l3p2, N3, l2p2, N2, l2f2, 64, s3, i2R3);
  lin1(s3, 64, CC3_0, t3, N3, 64, 1);
  lin1(t3, 64, CC3_1, l3f2_, N3, 64, 1);

  float* s3x  = alloc(N3*64);
  float* l3up = alloc(N3*64);
  sconv(l3p1, N3, l3p2, N3, l3f2_, 64, s3x, i2R3);
  // ModuleList shares one module -> both list entries identical; compute once
  lin(s3x, 64, l3f1_, 64, nullptr, 0, CC3_2, l3up, N3, 64, 1);

  // ---- decoder: level 3 -> 2 ----
  float* u2 = alloc(N2*64);
  sconv(l2p1, N2, l3p1, N3, l3up, 64, u2, i2R2);        // both upsamples identical
  float* t2c   = alloc(N2*96);
  float* t2d   = alloc(N2*96);
  float* s2c   = alloc(N2*96);
  float* l2f1b = alloc(N2*96);
  lin(u2, 64, u2, 64, l2f1, 64, CC2_3, t2c, N2, 96, 1); // concat(u2,u2,l2f1) 192ch
  lin1(t2c, 96, CC2_4, t2d, N2, 96, 1);
  sconv(l2p1, N2, l2p1, N2, t2d, 96, s2c, i2R2);
  lin1(s2c, 96, CC2_5, l2f1b, N2, 96, 1);

  // ---- decoder: level 2 -> 1 ----
  float* u1 = alloc(N1*96);
  sconv(l1p1, N1, l2p1, N2, l2f1b, 96, u1, i2R1);
  float* t1c   = alloc(N1*64);
  float* t1d   = alloc(N1*64);
  float* s1c   = alloc(N1*64);
  float* l1f1b = alloc(N1*64);
  lin(u1, 96, l1f1, 32, nullptr, 0, CC1_3, t1c, N1, 64, 1);
  lin1(t1c, 64, CC1_4, t1d, N1, 64, 1);
  sconv(l1p1, N1, l1p1, N1, t1d, 64, s1c, i2R1);
  lin1(s1c, 64, CC1_5, l1f1b, N1, 64, 1);

  // ---- decoder: level 1 -> 0 ----
  float* u0 = alloc(N0*64);
  sconv(pc1, N0, l1p1, N1, l1f1b, 64, u0, i2R0);
  float* t0c = alloc(N0*32);
  float* t0d = alloc(N0*32);
  float* s0c = alloc(N0*32);
  float* t0e = alloc(N0*32);
  lin(u0, 64, l0f1, 16, nullptr, 0, CC0_3, t0c, N0, 32, 1);
  lin1(t0c, 32, CC0_4, t0d, N0, 32, 1);
  sconv(pc1, N0, pc1, N0, t0d, 32, s0c, i2R0);
  lin1(s0c, 32, CC0_5, t0e, N0, 32, 1);

  // ---- refine head (no ReLU) -> d_out [4096,3] ----
  lin1(t0e, 32, REFINE, (float*)d_out, N0, 3, 0);
}
